// QLinearLayer_6768868459276
// MI455X (gfx1250) — 8-bit matrix operand order repaired and checked by reading, compiled, not run on hardware
//
#include <hip/hip_runtime.h>
#include <hip/hip_bf16.h>


#define IN_F   4096
#define OUT_F  11008
#define TOKENS 256
#define SEG0_END 3712
#define SEG1_END 3968
#define SEG2_END 4096

#define BROW_PITCH 80

typedef int v8i __attribute__((ext_vector_type(8)));
typedef int v4i __attribute__((ext_vector_type(4)));
typedef int v2i __attribute__((ext_vector_type(2)));

__global__ __launch_bounds__(256)
void quant_rows_kernel(const float* __restrict__ src,
                       const int*   __restrict__ reorder,
                       signed char* __restrict__ qout,
                       float*       __restrict__ sout)
{
    __shared__ float vals[IN_F];
    __shared__ float red[256];

    const int row = blockIdx.x;
    const int tid = threadIdx.x;
    const float* srow = src + (size_t)row * IN_F;

    for (int j = tid; j < IN_F; j += 256)
        vals[j] = srow[reorder[j]];
    __syncthreads();

    const int   seg_lo[3]   = {0, SEG0_END, SEG1_END};
    const int   seg_hi[3]   = {SEG0_END, SEG1_END, SEG2_END};
    const float seg_qmax[3] = {7.0f, 31.0f, 127.0f};

    #pragma unroll
    for (int seg = 0; seg < 3; ++seg) {
        const int s0 = seg_lo[seg], s1 = seg_hi[seg];
        const float qmax = seg_qmax[seg];

        float mx = 0.0f;
        for (int j = s0 + tid; j < s1; j += 256)
            mx = fmaxf(mx, fabsf(vals[j]));
        red[tid] = mx;
        __syncthreads();
        for (int o = 128; o > 0; o >>= 1) {
            if (tid < o) red[tid] = fmaxf(red[tid], red[tid + o]);
            __syncthreads();
        }
        float s = red[0] / qmax;
        if (s == 0.0f) s = 1.0f;
        if (tid == 0) sout[(size_t)row * 3 + seg] = s;

        for (int j = s0 + tid; j < s1; j += 256) {
            float q = rintf(fminf(fmaxf(vals[j] / s, -qmax), qmax));
            qout[(size_t)row * IN_F + j] = (signed char)q;
        }
        __syncthreads();
    }
}

__global__ __launch_bounds__(512)
void qgemm_wmma_kernel(const signed char* __restrict__ qx,
                       const float*       __restrict__ sx,
                       const signed char* __restrict__ qw,
                       const float*       __restrict__ sw,
                       const float*       __restrict__ bias,
                       float*             __restrict__ out)
{
    __shared__ signed char btile[2][64 * BROW_PITCH];

    const int tid  = threadIdx.x;
    const int lane = tid & 31;
    const int wave = tid >> 5;
    const int m0 = wave * 16;
    const int n0 = blockIdx.x * 64;

    const int lm = lane & 15;
    const int h  = lane >> 4;

    const int srow = tid >> 3;
    const int scol = (tid & 7) << 3;
    const signed char* gB = qw + (size_t)(n0 + srow) * IN_F + scol;
    const int sOff = srow * BROW_PITCH + scol;

    const signed char* arow = qx + (size_t)(m0 + lm) * IN_F + (h << 4);

    float facc[4][8];
    #pragma unroll
    for (int nn = 0; nn < 4; ++nn)
        #pragma unroll
        for (int r = 0; r < 8; ++r)
            facc[nn][r] = 0.0f;

    unsigned long long breg = *(const unsigned long long*)gB;

    auto run_seg = [&](int ks_beg, int ks_end, int seg) {
        v8i acc[4] = {};
        for (int ks = ks_beg; ks < ks_end; ++ks) {
            const int buf = ks & 1;
            *(unsigned long long*)(&btile[buf][sOff]) = breg;
            __syncthreads();
            if (ks + 1 < 64)
                breg = *(const unsigned long long*)(gB + ((ks + 1) << 6));

            const signed char* pa = arow + (ks << 6);
            v2i t0 = *(const v2i*)(pa +  0);
            v2i t1 = *(const v2i*)(pa +  8);
            v2i t2 = *(const v2i*)(pa + 32);
            v2i t3 = *(const v2i*)(pa + 40);
            v8i a;
            a[0] = t0[0]; a[1] = t0[1];
            a[2] = t1[0]; a[3] = t1[1];
            a[4] = t2[0]; a[5] = t2[1];
            a[6] = t3[0]; a[7] = t3[1];

            #pragma unroll
            for (int nn = 0; nn < 4; ++nn) {
                const signed char* pb = &btile[buf][(nn * 16 + lm) * BROW_PITCH + (h << 4)];
                v4i u0 = *(const v4i*)(pb +  0);
                v4i u1 = *(const v4i*)(pb + 32);
                v8i b;
                b[0] = u0[0]; b[1] = u0[1]; b[2] = u0[2]; b[3] = u0[3];
                b[4] = u1[0]; b[5] = u1[1]; b[6] = u1[2]; b[7] = u1[3];
                acc[nn] = __builtin_amdgcn_wmma_i32_16x16x64_iu8(
                    true, a, true, b, acc[nn], false, false);
            }
        }
        float sxr[8];
        #pragma unroll
        for (int r = 0; r < 8; ++r)
            sxr[r] = sx[(size_t)(m0 + (h << 3) + r) * 3 + seg];
        #pragma unroll
        for (int nn = 0; nn < 4; ++nn) {
            const float sb = sw[(size_t)(n0 + nn * 16 + lm) * 3 + seg];
            #pragma unroll
            for (int r = 0; r < 8; ++r)
                facc[nn][r] += (float)acc[nn][r] * (sxr[r] * sb);
        }
    };

    run_seg(0, 58, 0);
    run_seg(58, 62, 1);
    run_seg(62, 64, 2);

    #pragma unroll
    for (int nn = 0; nn < 4; ++nn) {
        const int n = n0 + nn * 16 + lm;
        const float bz = bias[n];
        #pragma unroll
        for (int r = 0; r < 8; ++r) {
            const int m = m0 + (h << 3) + r;
            out[(size_t)m * OUT_F + n] = facc[nn][r] + bz;
        }
    }
}

extern "C" void kernel_launch(void* const* d_in, const int* in_sizes, int n_in,
                              void* d_out, int out_size, void* d_ws, size_t ws_size,
                              hipStream_t stream) {
    const float* x       = (const float*)d_in[0];
    const float* W       = (const float*)d_in[1];
    const float* bias    = (const float*)d_in[2];
    const int*   reorder = (const int*)d_in[3];
    float* out = (float*)d_out;

    char* ws = (char*)d_ws;
    size_t off = 0;
    auto alloc = [&](size_t bytes) -> char* {
        char* p = ws + off;
        off = (off + bytes + 255) & ~(size_t)255;
        return p;
    };
    signed char* qw = (signed char*)alloc((size_t)OUT_F * IN_F);
    signed char* qx = (signed char*)alloc((size_t)TOKENS * IN_F);
    float*       sw = (float*)alloc((size_t)OUT_F * 3 * sizeof(float));
    float*       sx = (float*)alloc((size_t)TOKENS * 3 * sizeof(float));
    (void)ws_size; (void)in_sizes; (void)n_in; (void)out_size;

    quant_rows_kernel<<<TOKENS, 256, 0, stream>>>(x, reorder, qx, sx);
    quant_rows_kernel<<<OUT_F, 256, 0, stream>>>(W, reorder, qw, sw);

    qgemm_wmma_kernel<<<172, 512, 0, stream>>>(qx, sx, qw, sw, bias, out);
}
